// RelativeMultiHeadAttention_76501957476446
// MI455X (gfx1250) — matrix operand order repaired and checked by reading, not compiled, not run on hardware
//
#include <hip/hip_runtime.h>
#include <hip/hip_bf16.h>

// ---------------- problem constants ----------------
constexpr int Bc  = 2;
constexpr int Tc  = 512;
constexpr int CMc = 256;   // model dim
constexpr int CZc = 128;   // pair channels
constexpr int Hc  = 16;    // heads
constexpr int T2c = 2 * Tc - 1; // 1023 relative positions

typedef __attribute__((ext_vector_type(16))) _Float16 v16h;
typedef __attribute__((ext_vector_type(8)))  float    v8f;

// ---------------- CDNA5 WMMA (16x16x32 f16 -> f32 acc) ----------------
__device__ __forceinline__ v8f wmma16(v16h a, v16h b, v8f c) {
  // 8 args: (neg_a, A, neg_b, B, c_mod, C, reuse_a, reuse_b)
  return __builtin_amdgcn_wmma_f32_16x16x32_f16(false, a, false, b, (short)0, c,
                                                false, false);
}

// ---------------- CDNA5 async global->LDS copy (ASYNCcnt path) -------------
// Generic LDS pointers carry the LDS byte offset in addr[31:0] (ISA 10.2
// aperture table), so truncation gives the VDST value.
__device__ __forceinline__ unsigned lds_off(const void* p) {
  return (unsigned)(size_t)p;
}
__device__ __forceinline__ void async_copy_b128(unsigned lds_byte,
                                                const void* gaddr) {
  asm volatile("global_load_async_to_lds_b128 %0, %1, off"
               :
               : "v"(lds_byte), "v"((unsigned long long)(size_t)gaddr)
               : "memory");
}
__device__ __forceinline__ void wait_async0() {
  asm volatile("s_wait_asynccnt 0x0" ::: "memory");
}

// A-fragment (16x32, MxK): lane L holds row M=L&15; halves 0..7 -> K=ka..ka+7,
// halves 8..15 -> K=ka+16..ka+23, ka = (L>>4)*8.  (ISA 7.12.2)
template <bool GUARD>
__device__ __forceinline__ v16h loadA_f32(const float* __restrict__ src, int ld,
                                          int row0, int rowmax, int k0, int kmax) {
  const int lane = threadIdx.x & 31;
  const int row  = row0 + (lane & 15);
  const int ka   = k0 + ((lane >> 4) << 3);
  v16h a = {};
  if (!GUARD || row < rowmax) {
    const float* rp = src + (long)row * ld;
#pragma unroll
    for (int h = 0; h < 8; ++h) {
      int kA = ka + h, kB = ka + 16 + h;
      if (kA < kmax) a[h]     = (_Float16)rp[kA];
      if (kB < kmax) a[h + 8] = (_Float16)rp[kB];
    }
  }
  return a;
}

template <bool GUARD>
__device__ __forceinline__ v16h loadA_f16(const _Float16* __restrict__ src, int ld,
                                          int row0, int rowmax, int k0, int kmax) {
  const int lane = threadIdx.x & 31;
  const int row  = row0 + (lane & 15);
  const int ka   = k0 + ((lane >> 4) << 3);
  v16h a = {};
  if (!GUARD || row < rowmax) {
    const _Float16* rp = src + (long)row * ld;
#pragma unroll
    for (int h = 0; h < 8; ++h) {
      int kA = ka + h, kB = ka + 16 + h;
      if (kA < kmax) a[h]     = rp[kA];
      if (kB < kmax) a[h + 8] = rp[kB];
    }
  }
  return a;
}

// B-fragment (32x16, KxN): lane L holds column N=L&15; halves 0..7 -> K=kb..kb+7,
// halves 8..15 -> K=kb+16..kb+23, kb = (L>>4)*8.  Element (k,n) fetched from src[(n0+n)*strideN + k*strideK].
template <bool GUARD>
__device__ __forceinline__ v16h loadB_f16(const _Float16* __restrict__ src,
                                          long strideN, long strideK,
                                          int k0, int kmax, int n0, int nmax) {
  const int lane = threadIdx.x & 31;
  const int n  = n0 + (lane & 15);
  const int kb = k0 + ((lane >> 4) << 3);
  v16h f = {};
  if (!GUARD || n < nmax) {
    const _Float16* p = src + (long)n * strideN;
#pragma unroll
    for (int h = 0; h < 16; ++h) {
      int k = kb + (h < 8 ? h : h + 8);
      if (k < kmax) f[h] = p[(long)k * strideK];
    }
  }
  return f;
}

template <bool GUARD>
__device__ __forceinline__ v16h loadB_f32(const float* __restrict__ src,
                                          long strideN, long strideK,
                                          int k0, int kmax, int n0, int nmax) {
  const int lane = threadIdx.x & 31;
  const int n  = n0 + (lane & 15);
  const int kb = k0 + ((lane >> 4) << 3);
  v16h f = {};
  if (!GUARD || n < nmax) {
    const float* p = src + (long)n * strideN;
#pragma unroll
    for (int h = 0; h < 16; ++h) {
      int k = kb + (h < 8 ? h : h + 8);
      if (k < kmax) f[h] = (_Float16)p[(long)k * strideK];
    }
  }
  return f;
}

// ---------------- kernel 1: input projections (X @ W^T + bias [+ head bias]) ---
template <bool GUARD>
__global__ __launch_bounds__(128) void k_proj(
    const float* __restrict__ X, const float* __restrict__ W,
    const float* __restrict__ bias,
    const float* __restrict__ hb0, const float* __restrict__ hb1,
    _Float16* __restrict__ out0, _Float16* __restrict__ out1, int rows) {
  const int wave  = threadIdx.x >> 5;
  const int tileM = blockIdx.x * 4 + wave;
  const int row0  = tileM * 16;
  const int n0    = blockIdx.y * 16;
  if (GUARD && row0 >= rows) return;

  v8f acc = {};
  for (int k0 = 0; k0 < CMc; k0 += 32) {
    if (k0 + 32 < CMc) {  // hint next weight panel into cache
      __builtin_prefetch(&W[(long)(n0 + (threadIdx.x & 15)) * CMc + k0 + 32], 0, 1);
    }
    v16h a = loadA_f32<GUARD>(X, CMc, row0, rows, k0, CMc);
    v16h b = loadB_f32<false>(W, CMc, 1, k0, CMc, n0, CMc); // B[k][n] = W[n][k]
    acc = wmma16(a, b, acc);
  }
  const int lane = threadIdx.x & 31;
  const int n = n0 + (lane & 15);
  const float bb = bias ? bias[n] : 0.f;
  const float h0 = hb0 ? hb0[n] : 0.f;
  const float h1 = hb1 ? hb1[n] : 0.f;
#pragma unroll
  for (int r = 0; r < 8; ++r) {
    int m = row0 + r + ((lane >> 4) << 3);
    if (!GUARD || m < rows) {
      float v = acc[r] + bb;
      out0[(long)m * CMc + n] = (_Float16)(v + h0);
      if (out1) out1[(long)m * CMc + n] = (_Float16)(v + h1);
    }
  }
}

// ---------------- kernel 2: fused score ------------------------------------
// Per block: one (b, 16x16 (i,j) tile).
//  * z tile DMA'd into LDS once (async_to_lds, coalesced B128) in two 128-pair
//    chunks -> LN stats -> z-bias WMMA GEMM (K=128) all out of LDS
//  * per-head content WMMA + banded pos dot (rel-shift folded: col j-i+T-1)
//  * mask -> score (f32)
constexpr int ZPAD = 132; // f32 row stride: 16B aligned, 4-bank skew per pair
__global__ __launch_bounds__(256) void k_score(
    const _Float16* __restrict__ qu, const _Float16* __restrict__ qv,
    const _Float16* __restrict__ kh, const _Float16* __restrict__ ph,
    const float* __restrict__ z, const unsigned char* __restrict__ mask,
    const float* __restrict__ ln_g, const float* __restrict__ ln_b,
    const float* __restrict__ Wz, float* __restrict__ score) {
  __shared__ __align__(16) float s_z[128 * ZPAD];     // 66 KB staging chunk
  __shared__ float s_mu[256], s_rs[256];
  __shared__ float s_zb[256][Hc];                     // 256 pairs x 16 heads
  __shared__ float s_lng[CZc], s_lnb[CZc];
  __shared__ __align__(16) _Float16 s_qv[8][16 * 16]; // per-wave q(+v) tile
  __shared__ __align__(16) _Float16 s_ph[8][32 * 16]; // per-wave p band (31 rows)

  const int b  = blockIdx.z;
  const int i0 = blockIdx.y * 16;
  const int j0 = blockIdx.x * 16;
  const int wave = threadIdx.x >> 5;
  const int lane = threadIdx.x & 31;

  if (threadIdx.x < CZc) {
    s_lng[threadIdx.x] = ln_g[threadIdx.x];
    s_lnb[threadIdx.x] = ln_b[threadIdx.x];
  }

  // ---- z-bias: stage z in LDS (single HBM pass), stats, WMMA GEMM ----
  for (int ck = 0; ck < 2; ++ck) {
    const int pbase = ck * 128;
    // async DMA: 128 pair-rows (64KB), 16 x B128 per thread, fully coalesced
    {
      const unsigned zbase = lds_off(&s_z[0]);
#pragma unroll
      for (int r = 0; r < 16; ++r) {
        int q  = r * 256 + threadIdx.x;  // 0..4095 B128 transfers
        int pl = q >> 5;                 // local pair 0..127
        int ch = q & 31;                 // 16B chunk within 512B row
        int pp = pbase + pl;
        int ii = i0 + (pp >> 4), jj = j0 + (pp & 15);
        const char* g =
            (const char*)(z + (((long)b * Tc + ii) * Tc + jj) * CZc) + ch * 16;
        async_copy_b128(zbase + (unsigned)(pl * (ZPAD * 4) + ch * 16), g);
      }
      wait_async0();
    }
    __syncthreads();

    // LN stats for this chunk (one thread per pair, LDS-resident)
    if (threadIdx.x < 128) {
      const float* zp = &s_z[threadIdx.x * ZPAD];
      float s = 0.f, ss = 0.f;
#pragma unroll 4
      for (int c = 0; c < CZc; ++c) { float v = zp[c]; s += v; ss += v * v; }
      float mu = s * (1.f / CZc);
      float var = ss * (1.f / CZc) - mu * mu;
      s_mu[pbase + threadIdx.x] = mu;
      s_rs[pbase + threadIdx.x] = rsqrtf(var + 1e-5f);
    }
    __syncthreads();

    // z-bias GEMM: 8 row-tiles of (16 pairs x 16 heads), K = CZ = 128
    {
      const int rt = ck * 8 + wave;
      const int m  = lane & 15;
      const int pl = wave * 16 + m;          // local pair within chunk
      const int p  = rt * 16 + m;            // global pair
      const float mu = s_mu[p], rs = s_rs[p];
      const float* zp = &s_z[pl * ZPAD];
      v8f acc = {};
      for (int k0 = 0; k0 < CZc; k0 += 32) {
        const int ka = k0 + ((lane >> 4) << 3);
        v16h a = {};
#pragma unroll
        for (int h = 0; h < 8; ++h) {
          int c0 = ka + h, c1 = ka + 16 + h;  // always < CZ
          a[h]     = (_Float16)((zp[c0] - mu) * rs * s_lng[c0] + s_lnb[c0]);
          a[h + 8] = (_Float16)((zp[c1] - mu) * rs * s_lng[c1] + s_lnb[c1]);
        }
        v16h bf = loadB_f32<false>(Wz, 1, Hc, k0, CZc, 0, Hc); // B[k][n]=Wz[k][n]
        acc = wmma16(a, bf, acc);
      }
#pragma unroll
      for (int r = 0; r < 8; ++r) {
        int pw = rt * 16 + r + ((lane >> 4) << 3);
        s_zb[pw][lane & 15] = acc[r];
      }
    }
    __syncthreads();  // before s_z reuse / before consumers
  }

  // ---- per-head content (WMMA) + pos (banded dot) ----
  const int prow_base = j0 - i0 + (Tc - 16);  // band rows in [0,1022]
  for (int hi = 0; hi < 2; ++hi) {
    const int h = wave + hi * 8;
    __syncthreads();
    {
      // q(+v) tile: 512B = 1 B128 per lane
      {
        int m = lane >> 1, c = lane & 1;
        const char* g =
            (const char*)(qv + ((long)(b * Tc + i0 + m)) * CMc + h * 16) + c * 16;
        async_copy_b128(lds_off(&s_qv[wave][0]) + lane * 16, g);
      }
      // p band: 31 rows x 32B = 62 B128 (2 per lane, last 2 lanes do 1)
      const unsigned phb = lds_off(&s_ph[wave][0]);
      for (int t = lane; t < 62; t += 32) {
        int rr = t >> 1, cc = t & 1;
        const char* g =
            (const char*)(ph + ((long)(b * T2c + prow_base + rr)) * CMc + h * 16) +
            cc * 16;
        async_copy_b128(phb + t * 16, g);
      }
      wait_async0();
    }
    __syncthreads();

    // content: (q+u)[i-tile] x k[j-tile]^T, K = DH = 16 (zero-padded to 32)
    v16h af = loadA_f16<false>(qu + (long)b * Tc * CMc + h * 16, CMc, i0, Tc, 0, 16);
    v16h bf = loadB_f16<false>(kh + (long)b * Tc * CMc + h * 16, CMc, 1, 0, 16, j0, Tc);
    v8f czero = {};
    v8f cacc = wmma16(af, bf, czero);

    const int n = lane & 15;
    const int j = j0 + n;
    const bool mj = mask[b * Tc + j] != 0;
#pragma unroll
    for (int r = 0; r < 8; ++r) {
      const int m = r + ((lane >> 4) << 3);
      // pos term: (q+v)[i] . p[j-i+T-1]; local band row = n - m + 15 in [0,30]
      float pv = 0.f;
#pragma unroll
      for (int d = 0; d < 16; ++d)
        pv += (float)s_qv[wave][m * 16 + d] *
              (float)s_ph[wave][(n - m + 15) * 16 + d];
      float sc = (cacc[r] + pv) * 0.25f + s_zb[m * 16 + n][h];
      if (mj) sc = -10000.f;
      score[(((long)b * Hc + h) * Tc + (i0 + m)) * Tc + j] = sc;
    }
  }
}

// ---------------- kernel 3: row softmax (512 wide), attn -> f16 -------------
__global__ __launch_bounds__(256) void k_softmax(const float* __restrict__ score,
                                                 _Float16* __restrict__ attn) {
  __shared__ float red[8];
  const long row = blockIdx.x;                 // 0 .. B*H*T-1
  const float* sp = score + row * Tc;
  const int t = threadIdx.x;
  float v0 = sp[t], v1 = sp[t + 256];

  float mx = fmaxf(v0, v1);
#pragma unroll
  for (int off = 16; off > 0; off >>= 1) mx = fmaxf(mx, __shfl_xor(mx, off, 32));
  if ((t & 31) == 0) red[t >> 5] = mx;
  __syncthreads();
  float rowmax = red[0];
#pragma unroll
  for (int i = 1; i < 8; ++i) rowmax = fmaxf(rowmax, red[i]);
  __syncthreads();

  float e0 = __expf(v0 - rowmax), e1 = __expf(v1 - rowmax);
  float sm = e0 + e1;
#pragma unroll
  for (int off = 16; off > 0; off >>= 1) sm += __shfl_xor(sm, off, 32);
  if ((t & 31) == 0) red[t >> 5] = sm;
  __syncthreads();
  float tot = 0.f;
#pragma unroll
  for (int i = 0; i < 8; ++i) tot += red[i];
  float inv = 1.f / tot;

  attn[row * Tc + t]       = (_Float16)(e0 * inv);
  attn[row * Tc + t + 256] = (_Float16)(e1 * inv);
}

// ---------------- kernel 4: ctx = attn @ v (per head, K = T = 512) ----------
__global__ __launch_bounds__(128) void k_ctx(const _Float16* __restrict__ attn,
                                             const _Float16* __restrict__ vh,
                                             _Float16* __restrict__ ctx) {
  const int wave = threadIdx.x >> 5;
  const int it = blockIdx.x * 4 + wave;        // i-tile 0..31
  const int h = blockIdx.y, b = blockIdx.z;

  const _Float16* asrc = attn + (long)(b * Hc + h) * Tc * Tc;
  const _Float16* bsrc = vh + (long)b * Tc * CMc + h * 16;
  v8f acc = {};
  for (int k0 = 0; k0 < Tc; k0 += 32) {
    v16h a  = loadA_f16<false>(asrc, Tc, it * 16, Tc, k0, Tc);
    v16h bf = loadB_f16<false>(bsrc, 1, CMc, k0, Tc, 0, 16); // B[j][d]=v[b,j,h,d]
    acc = wmma16(a, bf, acc);
  }
  const int lane = threadIdx.x & 31;
  const int n = lane & 15;
#pragma unroll
  for (int r = 0; r < 8; ++r) {
    int m = it * 16 + r + ((lane >> 4) << 3);
    ctx[((long)(b * Tc + m)) * CMc + h * 16 + n] = (_Float16)acc[r];
  }
}

// ---------------- kernel 5: out = ctx @ Wo^T + bo (f32 output) --------------
__global__ __launch_bounds__(128) void k_out(const _Float16* __restrict__ ctxh,
                                             const float* __restrict__ Wo,
                                             const float* __restrict__ bo,
                                             float* __restrict__ out) {
  const int wave  = threadIdx.x >> 5;
  const int tileM = blockIdx.x * 4 + wave;
  const int row0  = tileM * 16;
  const int n0    = blockIdx.y * 16;

  v8f acc = {};
  for (int k0 = 0; k0 < CMc; k0 += 32) {
    if (k0 + 32 < CMc) {
      __builtin_prefetch(&Wo[(long)(n0 + (threadIdx.x & 15)) * CMc + k0 + 32], 0, 1);
    }
    v16h a = loadA_f16<false>(ctxh, CMc, row0, Bc * Tc, k0, CMc);
    v16h b = loadB_f32<false>(Wo, CMc, 1, k0, CMc, n0, CMc);
    acc = wmma16(a, b, acc);
  }
  const int lane = threadIdx.x & 31;
  const int n = n0 + (lane & 15);
  const float bb = bo[n];
#pragma unroll
  for (int r = 0; r < 8; ++r) {
    int m = row0 + r + ((lane >> 4) << 3);
    out[(long)m * CMc + n] = acc[r] + bb;
  }
}

// ---------------- host side -------------------------------------------------
extern "C" void kernel_launch(void* const* d_in, const int* in_sizes, int n_in,
                              void* d_out, int out_size, void* d_ws, size_t ws_size,
                              hipStream_t stream) {
  (void)in_sizes; (void)n_in; (void)out_size; (void)ws_size;

  const float* query = (const float*)d_in[0];
  const float* key   = (const float*)d_in[1];
  const float* value = (const float*)d_in[2];
  const float* pos   = (const float*)d_in[3];
  const float* z     = (const float*)d_in[4];
  const unsigned char* mask = (const unsigned char*)d_in[5]; // bool array
  const float* Wq = (const float*)d_in[6];
  const float* bq = (const float*)d_in[7];
  const float* Wk = (const float*)d_in[8];
  const float* bk = (const float*)d_in[9];
  const float* Wv = (const float*)d_in[10];
  const float* bv = (const float*)d_in[11];
  const float* Wp = (const float*)d_in[12];
  const float* u_bias = (const float*)d_in[13]; // (H,DH) flat = per-col bias
  const float* v_bias = (const float*)d_in[14];
  const float* Wo = (const float*)d_in[15];
  const float* bo = (const float*)d_in[16];
  const float* ln_g = (const float*)d_in[17];
  const float* ln_b = (const float*)d_in[18];
  const float* Wz = (const float*)d_in[19];
  float* out = (float*)d_out;

  // workspace carve-up (256B aligned)
  size_t off = 0;
  auto carve = [&](size_t bytes) {
    size_t o = off;
    off = (off + bytes + 255) & ~(size_t)255;
    return o;
  };
  char* ws = (char*)d_ws;
  _Float16* qu  = (_Float16*)(ws + carve((size_t)Bc * Tc * CMc * 2));
  _Float16* qv  = (_Float16*)(ws + carve((size_t)Bc * Tc * CMc * 2));
  _Float16* kh  = (_Float16*)(ws + carve((size_t)Bc * Tc * CMc * 2));
  _Float16* vh  = (_Float16*)(ws + carve((size_t)Bc * Tc * CMc * 2));
  _Float16* ph  = (_Float16*)(ws + carve((size_t)Bc * T2c * CMc * 2));
  _Float16* ctx = (_Float16*)(ws + carve((size_t)Bc * Tc * CMc * 2));
  float*    sc  = (float*)   (ws + carve((size_t)Bc * Hc * Tc * Tc * 4));
  _Float16* at  = (_Float16*)(ws + carve((size_t)Bc * Hc * Tc * Tc * 2));

  const dim3 blk128(128), blk256(256);

  // 1) projections: q (both u/v biased copies), k, v, p
  k_proj<false><<<dim3(16, 16), blk128, 0, stream>>>(query, Wq, bq, u_bias,
                                                     v_bias, qu, qv, Bc * Tc);
  k_proj<false><<<dim3(16, 16), blk128, 0, stream>>>(key, Wk, bk, nullptr,
                                                     nullptr, kh, nullptr, Bc * Tc);
  k_proj<false><<<dim3(16, 16), blk128, 0, stream>>>(value, Wv, bv, nullptr,
                                                     nullptr, vh, nullptr, Bc * Tc);
  k_proj<true><<<dim3(32, 16), blk128, 0, stream>>>(pos, Wp, nullptr, nullptr,
                                                    nullptr, ph, nullptr, Bc * T2c);

  // 2) fused score: async z staging + LN + z-bias GEMM + content WMMA + pos
  k_score<<<dim3(Tc / 16, Tc / 16, Bc), blk256, 0, stream>>>(
      qu, qv, kh, ph, z, mask, ln_g, ln_b, Wz, sc);

  // 3) softmax over j (row length 512)
  k_softmax<<<dim3(Bc * Hc * Tc), blk256, 0, stream>>>(sc, at);

  // 4) ctx = attn @ v
  k_ctx<<<dim3(Tc / 16 / 4, Hc, Bc), blk128, 0, stream>>>(at, vh, ctx);

  // 5) output projection
  k_out<<<dim3(16, 16), blk128, 0, stream>>>(ctx, Wo, bo, out);
}
